// RNNDecoder_5643587027288
// MI455X (gfx1250) — matrix operand order repaired and checked by reading, not compiled, not run on hardware
//
#include <hip/hip_runtime.h>

typedef __attribute__((ext_vector_type(16))) __bf16 v16bf;
typedef __attribute__((ext_vector_type(8)))  __bf16 v8bf;
typedef __attribute__((ext_vector_type(8)))  float  v8f;

#define HORIZON 12
#define BATCH   16384
#define TILE_B  32      // batch rows per workgroup (2 WMMA M-tiles)
#define NTHREADS 256    // 8 waves (wave32)

// ---------- helpers ----------
__device__ __forceinline__ __bf16 f2bf(float f) {
  unsigned u = __builtin_bit_cast(unsigned, f);
  u += 0x7FFFu + ((u >> 16) & 1u);            // round-to-nearest-even
  unsigned short s = (unsigned short)(u >> 16);
  return __builtin_bit_cast(__bf16, s);
}
__device__ __forceinline__ float sigmoidf_(float x) {
  return 1.0f / (1.0f + __expf(-x));
}
__device__ __forceinline__ float tanhf_(float x) {
  x = fminf(fmaxf(x, -15.f), 15.f);
  float e = __expf(2.f * x);
  return (e - 1.f) / (e + 1.f);
}
__device__ __forceinline__ v8f vzero8() {
  v8f v = {0.f,0.f,0.f,0.f,0.f,0.f,0.f,0.f};
  return v;
}
__device__ __forceinline__ v8f wmma_bf16(v16bf a, v16bf b, v8f c) {
  return __builtin_amdgcn_wmma_f32_16x16x32_bf16(false, a, false, b, (short)0, c, false, false);
}

// A fragment (16x32 bf16): lane holds row M = m0 + (lane%16);
// half = lane/16: K = {kc*32 + half*16 .. +15}
__device__ __forceinline__ v16bf load_a(const __bf16* buf, int m0, int kc, int lm, int half) {
  int base = (m0 + lm) * 256 + kc * 32 + half * 16;
  union { v16bf v; v8bf h[2]; } a;
  a.h[0] = *reinterpret_cast<const v8bf*>(buf + base);
  a.h[1] = *reinterpret_cast<const v8bf*>(buf + base + 8);
  return a.v;
}
// B fragment (32x16 bf16): lane holds column N = n0 + (lane%16),
// 16 consecutive K values starting at kc*32 + half*16 -> one contiguous 32B chunk
// of row-major W[n][k].
__device__ __forceinline__ v16bf load_b(const __bf16* w, int n0, int kc, int lm, int half) {
  int off = (n0 + lm) * 256 + kc * 32 + half * 16;
  return *reinterpret_cast<const v16bf*>(w + off);
}

// ---------- weight pre-pack: fp32 -> bf16, fused biases ----------
__global__ void prepack_kernel(const float* __restrict__ Wih_f, const float* __restrict__ Whh_f,
                               const float* __restrict__ bih,   const float* __restrict__ bhh,
                               const float* __restrict__ Wlin_f,const float* __restrict__ blin_f,
                               __bf16* __restrict__ Wih, __bf16* __restrict__ Whh,
                               __bf16* __restrict__ Wlin,
                               float* __restrict__ brz, float* __restrict__ bin,
                               float* __restrict__ bhn, float* __restrict__ blin) {
  int idx = blockIdx.x * blockDim.x + threadIdx.x;
  const int NW = 2 * 768 * 256;
  if (idx < NW) {
    Wih[idx] = f2bf(Wih_f[idx]);
    Whh[idx] = f2bf(Whh_f[idx]);
  }
  if (idx < 256 * 256) Wlin[idx] = f2bf(Wlin_f[idx]);
  if (idx < 2 * 512) {                 // r,z biases fused (i + h)
    int l = idx >> 9, i = idx & 511;
    brz[idx] = bih[l * 768 + i] + bhh[l * 768 + i];
  }
  if (idx < 2 * 256) {                 // n-gate biases kept separate
    int l = idx >> 8, j = idx & 255;
    bin[idx] = bih[l * 768 + 512 + j];
    bhn[idx] = bhh[l * 768 + 512 + j];
  }
  if (idx < 256) blin[idx] = blin_f[idx];
}

// ---------- fused autoregressive GRU decoder ----------
// Each wave owns gate columns [wv*32, wv*32+32); the two 16-wide N-tiles are
// processed SEQUENTIALLY so only 64 accumulator VGPRs are live at a time
// (4 gates x 2 M-tiles), keeping the wave under 256 VGPRs so two workgroups
// (4 waves/SIMD) can co-reside per WGP alongside the 112 KB LDS footprint.
__global__ __launch_bounds__(NTHREADS)
void gru_decoder_kernel(const float* __restrict__ y0,
                        const float* __restrict__ h0,
                        const __bf16* __restrict__ Wih,   // [2][768][256] bf16 row-major
                        const __bf16* __restrict__ Whh,   // [2][768][256]
                        const __bf16* __restrict__ Wlin,  // [256][256]
                        const float* __restrict__ brz,    // [2][512]
                        const float* __restrict__ bin,    // [2][256]
                        const float* __restrict__ bhn,    // [2][256]
                        const float* __restrict__ blin,   // [256]
                        float* __restrict__ out) {        // [12][B][256] f32
  __shared__ __bf16 xb[TILE_B * 256];          // 16 KB  current input y_{t-1}
  __shared__ __bf16 hbuf[2][TILE_B * 256];     // 32 KB  h per layer (bf16 for WMMA A)
  __shared__ float  hf[2][TILE_B * 256];       // 64 KB  h per layer (f32 for blend)
                                               // total 112 KB -> 2 WGs per WGP

  const int tid  = threadIdx.x;
  const int lane = tid & 31;
  const int wv   = tid >> 5;       // 0..7
  const int lm   = lane & 15;
  const int half = lane >> 4;
  const int gb0  = blockIdx.x * TILE_B;

  // ---- stage initial state
  for (int i = tid; i < TILE_B * 256; i += NTHREADS) {
    int r = i >> 8, c = i & 255;
    float xv  = y0[(size_t)(gb0 + r) * 256 + c];
    float h0v = h0[(size_t)0 * BATCH * 256 + (size_t)(gb0 + r) * 256 + c];
    float h1v = h0[(size_t)1 * BATCH * 256 + (size_t)(gb0 + r) * 256 + c];
    xb[i] = f2bf(xv);
    hf[0][i] = h0v;          hf[1][i] = h1v;
    hbuf[0][i] = f2bf(h0v);  hbuf[1][i] = f2bf(h1v);
  }
  __syncthreads();

  for (int t = 0; t < HORIZON; ++t) {
    for (int l = 0; l < 2; ++l) {
      const __bf16* xsrc = (l == 0) ? xb : hbuf[0];
      const __bf16* wih  = Wih + (size_t)l * 768 * 256;
      const __bf16* whh  = Whh + (size_t)l * 768 * 256;

      float hn[2][2][8];   // deferred h' values [N-tile][M-tile][row]

      #pragma unroll
      for (int i = 0; i < 2; ++i) {
        const int j0 = wv * 32 + i * 16;

        // accumulators for THIS N-tile only: 4 gates x 2 M-tiles = 64 VGPRs
        v8f aR[2], aZ[2], aI[2], aH[2];
        #pragma unroll
        for (int mi = 0; mi < 2; ++mi) {
          aR[mi] = vzero8(); aZ[mi] = vzero8();
          aI[mi] = vzero8(); aH[mi] = vzero8();
        }

        // ---- x @ Wih^T contributions (r, z, i_n)
        #pragma unroll
        for (int kc = 0; kc < 8; ++kc) {
          v16bf a0 = load_a(xsrc, 0, kc, lm, half);
          v16bf a1 = load_a(xsrc, 16, kc, lm, half);
          v16bf br = load_b(wih, j0,       kc, lm, half);
          v16bf bz = load_b(wih, 256 + j0, kc, lm, half);
          v16bf bn = load_b(wih, 512 + j0, kc, lm, half);
          aR[0] = wmma_bf16(a0, br, aR[0]);
          aR[1] = wmma_bf16(a1, br, aR[1]);
          aZ[0] = wmma_bf16(a0, bz, aZ[0]);
          aZ[1] = wmma_bf16(a1, bz, aZ[1]);
          aI[0] = wmma_bf16(a0, bn, aI[0]);
          aI[1] = wmma_bf16(a1, bn, aI[1]);
        }
        // ---- h @ Whh^T contributions (r, z, h_n)
        #pragma unroll
        for (int kc = 0; kc < 8; ++kc) {
          v16bf a0 = load_a(hbuf[l], 0, kc, lm, half);
          v16bf a1 = load_a(hbuf[l], 16, kc, lm, half);
          v16bf br = load_b(whh, j0,       kc, lm, half);
          v16bf bz = load_b(whh, 256 + j0, kc, lm, half);
          v16bf bn = load_b(whh, 512 + j0, kc, lm, half);
          aR[0] = wmma_bf16(a0, br, aR[0]);
          aR[1] = wmma_bf16(a1, br, aR[1]);
          aZ[0] = wmma_bf16(a0, bz, aZ[0]);
          aZ[1] = wmma_bf16(a1, bz, aZ[1]);
          aH[0] = wmma_bf16(a0, bn, aH[0]);
          aH[1] = wmma_bf16(a1, bn, aH[1]);
        }

        // ---- gate nonlinearity in-register; defer the LDS write of h'
        {
          int j = j0 + lm;                   // 0..255
          float br_ = brz[l * 512 + j];
          float bz_ = brz[l * 512 + 256 + j];
          float bi_ = bin[l * 256 + j];
          float bh_ = bhn[l * 256 + j];
          #pragma unroll
          for (int mi = 0; mi < 2; ++mi)
            #pragma unroll
            for (int v = 0; v < 8; ++v) {
              int m   = mi * 16 + half * 8 + v;  // C/D layout: lanes16-31 => M+8
              int idx = m * 256 + j;
              float r  = sigmoidf_(aR[mi][v] + br_);
              float z  = sigmoidf_(aZ[mi][v] + bz_);
              float n  = tanhf_(aI[mi][v] + bi_ + r * (aH[mi][v] + bh_));
              hn[i][mi][v] = (1.f - z) * n + z * hf[l][idx];
            }
        }
      } // N-tile loop

      __syncthreads();   // all waves done READING hbuf[l]/xsrc before overwrite

      // ---- commit h' to LDS (f32 + bf16)
      #pragma unroll
      for (int i = 0; i < 2; ++i) {
        int j = wv * 32 + i * 16 + lm;
        #pragma unroll
        for (int mi = 0; mi < 2; ++mi)
          #pragma unroll
          for (int v = 0; v < 8; ++v) {
            int m   = mi * 16 + half * 8 + v;
            int idx = m * 256 + j;
            float hv = hn[i][mi][v];
            hf[l][idx]   = hv;
            hbuf[l][idx] = f2bf(hv);
          }
      }
      __syncthreads();
    } // layers

    // ===== Linear head: y_t = h1 @ Wlin^T + b_lin; feed back as x =====
    {
      v8f acc[2][2];
      #pragma unroll
      for (int mi = 0; mi < 2; ++mi)
        #pragma unroll
        for (int i = 0; i < 2; ++i) acc[mi][i] = vzero8();

      #pragma unroll
      for (int kc = 0; kc < 8; ++kc) {
        v16bf a0 = load_a(hbuf[1], 0, kc, lm, half);
        v16bf a1 = load_a(hbuf[1], 16, kc, lm, half);
        #pragma unroll
        for (int i = 0; i < 2; ++i) {
          int n0 = (wv * 2 + i) * 16;
          v16bf b = load_b(Wlin, n0, kc, lm, half);
          acc[0][i] = wmma_bf16(a0, b, acc[0][i]);
          acc[1][i] = wmma_bf16(a1, b, acc[1][i]);
        }
      }
      #pragma unroll
      for (int i = 0; i < 2; ++i) {
        int j = (wv * 2 + i) * 16 + lm;
        float bv = blin[j];
        #pragma unroll
        for (int mi = 0; mi < 2; ++mi)
          #pragma unroll
          for (int v = 0; v < 8; ++v) {
            int m = mi * 16 + half * 8 + v;
            float yv = acc[mi][i][v] + bv;
            // nontemporal: output is write-once; keep L2 for the hot weight set
            __builtin_nontemporal_store(
                yv, &out[(size_t)t * BATCH * 256 + (size_t)(gb0 + m) * 256 + j]);
            xb[m * 256 + j] = f2bf(yv);   // next-step input
          }
      }
      __syncthreads();
    }
  } // t
}

// ---------- launcher ----------
extern "C" void kernel_launch(void* const* d_in, const int* in_sizes, int n_in,
                              void* d_out, int out_size, void* d_ws, size_t ws_size,
                              hipStream_t stream) {
  (void)in_sizes; (void)n_in; (void)out_size; (void)ws_size;
  const float* y0   = (const float*)d_in[0];
  const float* h0   = (const float*)d_in[1];
  const float* Wih  = (const float*)d_in[2];
  const float* Whh  = (const float*)d_in[3];
  const float* bih  = (const float*)d_in[4];
  const float* bhh  = (const float*)d_in[5];
  const float* Wlin = (const float*)d_in[6];
  const float* blin = (const float*)d_in[7];

  // workspace layout (bytes), all offsets 1KB-aligned
  char* ws = (char*)d_ws;
  __bf16* wih_b  = (__bf16*)(ws + 0);        // 786432 B
  __bf16* whh_b  = (__bf16*)(ws + 786432);   // 786432 B
  __bf16* wlin_b = (__bf16*)(ws + 1572864);  // 131072 B
  float*  brz    = (float*)(ws + 1703936);   // 4096 B
  float*  binp   = (float*)(ws + 1708032);   // 2048 B
  float*  bhnp   = (float*)(ws + 1710080);   // 2048 B
  float*  blinp  = (float*)(ws + 1712128);   // 1024 B

  prepack_kernel<<<(2 * 768 * 256 + 255) / 256, 256, 0, stream>>>(
      Wih, Whh, bih, bhh, Wlin, blin,
      wih_b, whh_b, wlin_b, brz, binp, bhnp, blinp);

  gru_decoder_kernel<<<BATCH / TILE_B, NTHREADS, 0, stream>>>(
      y0, h0, wih_b, whh_b, wlin_b, brz, binp, bhnp, blinp, (float*)d_out);
}
